// MultiHeadAttention_50706383896888
// MI455X (gfx1250) — compile-verified, matrix operand order checked by reading, not run on hardware
//
#include <hip/hip_runtime.h>
#include <hip/hip_bf16.h>

// ---------------------------------------------------------------------------
// MHA (GQA, RoPE, causal) for MI455X / gfx1250, wave32 + WMMA bf16.
//   B=2, S=2048, DIM=2048, H=32, KVH=8, HD=64, theta=1e4
// Pipeline:
//   1. x -> bf16;  w_qkv, w_proj -> bf16 transposed [N,K]
//   2. qkv = x @ w_qkv        (WMMA bf16 GEMM, f32 out)
//   3. RoPE+pack q [B,H,S,HD], k [B,KVH,S,HD], v^T [B,KVH,HD,S]  (bf16)
//   4. flash attention, transposed layout: S^T = K*Q^T, O^T = V^T*P^T
//   5. out = attn @ w_proj + b_proj  (WMMA bf16 GEMM, f32 out)
// ---------------------------------------------------------------------------

#define DIMC   2048
#define HEADS  32
#define KVHD   8
#define HD     64
#define BATCH  2
#define SEQ    2048
#define QKVN   3072      // H*HD + 2*KVH*HD
#define QSIZE  2048
#define KSIZE  512
#define GQA    4         // HEADS / KVHD

typedef __attribute__((ext_vector_type(16))) __bf16 bf16x16;
typedef __attribute__((ext_vector_type(8)))  float  f32x8;
typedef __attribute__((ext_vector_type(4)))  float  f32x4;

union Frag {                 // one WMMA 16-bit A/B fragment (8 VGPRs / lane)
  bf16x16        v;
  f32x4          q[2];
  unsigned short h[16];
};

__device__ __forceinline__ unsigned short f2bf(float f) {  // f32 -> bf16 RNE
  unsigned u = __float_as_uint(f);
  u += 0x7fffu + ((u >> 16) & 1u);
  return (unsigned short)(u >> 16);
}

// --------------------------- elementwise helpers ---------------------------

__global__ __launch_bounds__(256) void k_f32_to_bf16(
    const float* __restrict__ in, unsigned short* __restrict__ out, int n) {
  int i = blockIdx.x * 256 + threadIdx.x;
  if (i < n) out[i] = f2bf(in[i]);
}

// w [K,N] f32 row-major -> wt [N,K] bf16 row-major
__global__ __launch_bounds__(256) void k_transpose_bf16(
    const float* __restrict__ w, unsigned short* __restrict__ wt, int K, int N) {
  int i = blockIdx.x * 256 + threadIdx.x;
  if (i >= K * N) return;
  int k = i % K, n = i / K;
  wt[i] = f2bf(w[(size_t)k * N + n]);
}

// RoPE + repack: qkv f32 [B,S,QKVN] slice -> out bf16 [B,nheads,S,HD]
__global__ __launch_bounds__(256) void k_rope_pack(
    const float* __restrict__ qkv, unsigned short* __restrict__ out,
    int nheads, int col_off) {
  int i = blockIdx.x * 256 + threadIdx.x;
  int total = BATCH * nheads * SEQ * (HD / 2);
  if (i >= total) return;
  int d = i & 31; int t = i >> 5;
  int s = t % SEQ; t /= SEQ;
  int h = t % nheads; int b = t / nheads;
  const float* src = qkv + ((size_t)(b * SEQ + s)) * QKVN + col_off + h * HD;
  float x1 = src[d], x2 = src[d + 32];
  // inv_freq = theta^(-2d/HD) = exp(-d * ln(1e4)/32)
  float ang = (float)s * __expf(-(float)d * 0.28782313662425575f);
  float c = cosf(ang), sn = sinf(ang);
  unsigned short* dst = out + (((size_t)(b * nheads + h)) * SEQ + s) * HD;
  dst[d]      = f2bf(x1 * c - x2 * sn);
  dst[d + 32] = f2bf(x2 * c + x1 * sn);
}

// V slice of qkv -> v^T bf16 [B,KVH,HD,S]
__global__ __launch_bounds__(256) void k_vt_pack(
    const float* __restrict__ qkv, unsigned short* __restrict__ vt) {
  int i = blockIdx.x * 256 + threadIdx.x;
  const int total = BATCH * KVHD * HD * SEQ;
  if (i >= total) return;
  int s = i % SEQ; int t = i / SEQ;
  int hd = t % HD; t /= HD;
  int kv = t % KVHD; int b = t / KVHD;
  float v = qkv[((size_t)(b * SEQ + s)) * QKVN + QSIZE + KSIZE + kv * HD + hd];
  vt[i] = f2bf(v);
}

// ------------------------------- WMMA GEMM ---------------------------------
// C[M,N] f32 = A[M,K]bf16 @ BT[N,K]bf16^T (+bias). Block=128 thr = 4 waves,
// block tile 128x128, wave tile 64x64 (16 wmma/k-step). L2-resident, no LDS.

__global__ __launch_bounds__(128) void k_gemm_bf16(
    const unsigned short* __restrict__ A,
    const unsigned short* __restrict__ BT,
    const float* __restrict__ bias,
    float* __restrict__ C,
    int M, int N, int K) {
  const int lane  = threadIdx.x & 31;
  const int wave  = threadIdx.x >> 5;
  const int lrow  = lane & 15;
  const int khalf = lane >> 4;
  const int nb = N >> 7;
  const int m0 = (blockIdx.x / nb) * 128 + (wave >> 1) * 64;
  const int n0 = (blockIdx.x % nb) * 128 + (wave & 1) * 64;

  f32x8 acc[4][4] = {};
  for (int k0 = 0; k0 < K; k0 += 32) {
    Frag a[4], b[4];
#pragma unroll
    for (int i = 0; i < 4; ++i) {
      const unsigned short* pa = A + (size_t)(m0 + i * 16 + lrow) * K + (k0 + khalf * 8);
      a[i].q[0] = *reinterpret_cast<const f32x4*>(pa);
      a[i].q[1] = *reinterpret_cast<const f32x4*>(pa + 16);
      const unsigned short* pb = BT + (size_t)(n0 + i * 16 + lrow) * K + (k0 + khalf * 8);
      b[i].q[0] = *reinterpret_cast<const f32x4*>(pb);
      b[i].q[1] = *reinterpret_cast<const f32x4*>(pb + 16);
    }
#pragma unroll
    for (int i = 0; i < 4; ++i)
#pragma unroll
      for (int j = 0; j < 4; ++j)
        acc[i][j] = __builtin_amdgcn_wmma_f32_16x16x32_bf16(
            false, a[i].v, false, b[j].v, (short)0, acc[i][j], false, false);
  }
#pragma unroll
  for (int j = 0; j < 4; ++j) {
    const int ncol = n0 + j * 16 + lrow;
    const float bv = bias ? bias[ncol] : 0.0f;
#pragma unroll
    for (int i = 0; i < 4; ++i)
#pragma unroll
      for (int r = 0; r < 8; ++r)
        C[(size_t)(m0 + i * 16 + khalf * 8 + r) * N + ncol] = acc[i][j][r] + bv;
  }
}

// ---------------------------- flash attention ------------------------------
// One wave per (b, head, 16-query tile). Transposed layout:
//   S^T[key,q]  = K_tile(A) x Q^T(B)   -> lane = q column, regs = key rows
//   O^T[hd,q]  += V^T_tile(A) x P^T(B) -> P^T B-fragment is lane-local repack
// Online softmax stats are per-lane; one shfl_xor(16) merges key halves.

__global__ __launch_bounds__(256) void k_attn(
    const unsigned short* __restrict__ Q,   // [B,H,S,HD]
    const unsigned short* __restrict__ Kc,  // [B,KVH,S,HD]
    const unsigned short* __restrict__ VT,  // [B,KVH,HD,S]
    unsigned short* __restrict__ O) {       // [B,S,DIM]
  const int lane  = threadIdx.x & 31;
  const int lrow  = lane & 15;
  const int khalf = lane >> 4;
  int gw = blockIdx.x * 8 + (threadIdx.x >> 5);
  const int NQT = SEQ / 16;
  const int qt = gw % NQT; gw /= NQT;
  const int h = gw % HEADS; const int b = gw / HEADS;
  const int kv = h / GQA;
  const int q0 = qt * 16;
  const int qrow = q0 + lrow;

  // Q^T B-fragments (2 halves of HD=64), contiguous per-lane loads
  const unsigned short* qp = Q + (((size_t)(b * HEADS + h)) * SEQ + qrow) * HD + khalf * 8;
  Frag qf0, qf1;
  qf0.q[0] = *reinterpret_cast<const f32x4*>(qp);
  qf0.q[1] = *reinterpret_cast<const f32x4*>(qp + 16);
  qf1.q[0] = *reinterpret_cast<const f32x4*>(qp + 32);
  qf1.q[1] = *reinterpret_cast<const f32x4*>(qp + 48);

  const unsigned short* kb_base = Kc + ((size_t)(b * KVHD + kv)) * SEQ * HD;
  const unsigned short* vt_base = VT + ((size_t)(b * KVHD + kv)) * HD * SEQ;

  f32x8 oacc[4] = {};
  float mrun = -3.0e38f, lrun = 0.0f;

  for (int kb = 0; kb < q0 + 16; kb += 32) {
    f32x8 s0 = {}, s1 = {};
#pragma unroll
    for (int hk = 0; hk < 2; ++hk) {
      Frag ka, kc;
      const unsigned short* kp0 = kb_base + (size_t)(kb + lrow) * HD + hk * 32 + khalf * 8;
      ka.q[0] = *reinterpret_cast<const f32x4*>(kp0);
      ka.q[1] = *reinterpret_cast<const f32x4*>(kp0 + 16);
      const unsigned short* kp1 = kb_base + (size_t)(kb + 16 + lrow) * HD + hk * 32 + khalf * 8;
      kc.q[0] = *reinterpret_cast<const f32x4*>(kp1);
      kc.q[1] = *reinterpret_cast<const f32x4*>(kp1 + 16);
      const bf16x16 qv = (hk == 0) ? qf0.v : qf1.v;
      s0 = __builtin_amdgcn_wmma_f32_16x16x32_bf16(false, ka.v, false, qv, (short)0, s0, false, false);
      s1 = __builtin_amdgcn_wmma_f32_16x16x32_bf16(false, kc.v, false, qv, (short)0, s1, false, false);
    }
    // scale + causal mask + running max
    float mnew = mrun;
#pragma unroll
    for (int r = 0; r < 8; ++r) {
      int key0 = kb + khalf * 8 + r;
      s0[r] = (key0      <= qrow) ? s0[r] * 0.125f : -3.0e38f;
      s1[r] = (key0 + 16 <= qrow) ? s1[r] * 0.125f : -3.0e38f;
      mnew = fmaxf(mnew, fmaxf(s0[r], s1[r]));
    }
    mnew = fmaxf(mnew, __shfl_xor(mnew, 16, 32));
    const float corr = __expf(mrun - mnew);
    lrun *= corr;
#pragma unroll
    for (int r = 0; r < 8; ++r) {
      oacc[0][r] *= corr; oacc[1][r] *= corr; oacc[2][r] *= corr; oacc[3][r] *= corr;
    }
    // exponentiate and repack P^T into its B-fragment (pure lane-local)
    Frag pf;
    float psum = 0.0f;
#pragma unroll
    for (int r = 0; r < 8; ++r) {
      float p0 = __expf(s0[r] - mnew);
      float p1 = __expf(s1[r] - mnew);
      psum += p0 + p1;
      pf.h[r]     = f2bf(p0);
      pf.h[r + 8] = f2bf(p1);
    }
    lrun += psum;
    mrun = mnew;
    // O^T += V^T x P^T over 32 keys, 4 hd-tiles
#pragma unroll
    for (int t = 0; t < 4; ++t) {
      const unsigned short* vp = vt_base + (size_t)(t * 16 + lrow) * SEQ + kb + khalf * 8;
      Frag va;
      va.q[0] = *reinterpret_cast<const f32x4*>(vp);
      va.q[1] = *reinterpret_cast<const f32x4*>(vp + 16);
      oacc[t] = __builtin_amdgcn_wmma_f32_16x16x32_bf16(
          false, va.v, false, pf.v, (short)0, oacc[t], false, false);
    }
  }
  lrun += __shfl_xor(lrun, 16, 32);
  const float inv = 1.0f / lrun;
  // O^T lane = column q; reg r of tile t is hd = t*16 + khalf*8 + r
  unsigned short* op = O + ((size_t)(b * SEQ + qrow)) * DIMC + h * HD + khalf * 8;
#pragma unroll
  for (int t = 0; t < 4; ++t)
#pragma unroll
    for (int r = 0; r < 8; r += 2) {
      unsigned pack = (unsigned)f2bf(oacc[t][r] * inv) |
                      ((unsigned)f2bf(oacc[t][r + 1] * inv) << 16);
      *reinterpret_cast<unsigned*>(op + t * 16 + r) = pack;
    }
}

// ------------------------------- launcher ----------------------------------

extern "C" void kernel_launch(void* const* d_in, const int* in_sizes, int n_in,
                              void* d_out, int out_size, void* d_ws, size_t ws_size,
                              hipStream_t stream) {
  (void)in_sizes; (void)n_in; (void)out_size; (void)ws_size;
  const float* x      = (const float*)d_in[0];
  const float* w_qkv  = (const float*)d_in[1];
  const float* w_proj = (const float*)d_in[2];
  const float* b_proj = (const float*)d_in[3];
  float* out = (float*)d_out;

  char* ws = (char*)d_ws;
  size_t off = 0;
  auto take = [&](size_t bytes) {
    char* p = ws + off;
    off += (bytes + 255) & ~(size_t)255;
    return p;
  };
  unsigned short* xb   = (unsigned short*)take((size_t)BATCH * SEQ * DIMC * 2);
  unsigned short* wqT  = (unsigned short*)take((size_t)QKVN * DIMC * 2);
  unsigned short* wpT  = (unsigned short*)take((size_t)DIMC * DIMC * 2);
  float*          qkv  = (float*)take((size_t)BATCH * SEQ * QKVN * 4);
  unsigned short* qb   = (unsigned short*)take((size_t)BATCH * HEADS * SEQ * HD * 2);
  unsigned short* kbuf = (unsigned short*)take((size_t)BATCH * KVHD * SEQ * HD * 2);
  unsigned short* vtb  = (unsigned short*)take((size_t)BATCH * KVHD * HD * SEQ * 2);
  unsigned short* ab   = (unsigned short*)take((size_t)BATCH * SEQ * DIMC * 2);

  int n;
  n = BATCH * SEQ * DIMC;
  k_f32_to_bf16<<<(n + 255) / 256, 256, 0, stream>>>(x, xb, n);
  n = DIMC * QKVN;
  k_transpose_bf16<<<(n + 255) / 256, 256, 0, stream>>>(w_qkv, wqT, DIMC, QKVN);
  n = DIMC * DIMC;
  k_transpose_bf16<<<(n + 255) / 256, 256, 0, stream>>>(w_proj, wpT, DIMC, DIMC);

  {
    int M = BATCH * SEQ, N = QKVN, K = DIMC;
    k_gemm_bf16<<<(M / 128) * (N / 128), 128, 0, stream>>>(xb, wqT, nullptr, qkv, M, N, K);
  }
  n = BATCH * HEADS * SEQ * (HD / 2);
  k_rope_pack<<<(n + 255) / 256, 256, 0, stream>>>(qkv, qb, HEADS, 0);
  n = BATCH * KVHD * SEQ * (HD / 2);
  k_rope_pack<<<(n + 255) / 256, 256, 0, stream>>>(qkv, kbuf, KVHD, QSIZE);
  n = BATCH * KVHD * HD * SEQ;
  k_vt_pack<<<(n + 255) / 256, 256, 0, stream>>>(qkv, vtb);

  {
    int waves = BATCH * HEADS * (SEQ / 16);     // 8192 waves
    k_attn<<<waves / 8, 256, 0, stream>>>(qb, kbuf, vtb, ab);
  }
  {
    int M = BATCH * SEQ, N = DIMC, K = DIMC;
    k_gemm_bf16<<<(M / 128) * (N / 128), 128, 0, stream>>>(ab, wpT, b_proj, out, M, N, K);
  }
}
